// MultiHeadAttention_21363167330869
// MI455X (gfx1250) — compile-verified, matrix operand order checked by reading, not run on hardware
//
#include <hip/hip_runtime.h>
#include <hip/hip_bf16.h>
#include <math.h>

// ---------------- problem constants ----------------
#define E_DIM   1024
#define H_HEADS 16
#define D_HEAD  64
#define B_SZ    4
#define N_SEQ   2048
#define M_ROWS  (B_SZ * N_SEQ)        // 8192
#define QKV_N   (3 * E_DIM)           // 3072

typedef __attribute__((ext_vector_type(16))) __bf16 v16bf;
typedef __attribute__((ext_vector_type(8)))  float  v8f;

// ---------------------------------------------------------------------------
// Fragment loader following CDNA5 ISA 16-bit A/B VGPR layout (CDNA5 ISA):
//   lane half g = lane>>4; operand dword i holds K-pair at dword offset
//   (i < 4 ? i : i + 4) + 4*g within a 32-element (16-dword) K-run.
// ---------------------------------------------------------------------------
__device__ __forceinline__ v16bf ld_frag32(const uint32_t* p, int g) {
    union { v16bf v; uint32_t u[8]; } r;
#pragma unroll
    for (int i = 0; i < 8; ++i) {
        int d = (i < 4 ? i : i + 4) + 4 * g;
        r.u[i] = p[d];
    }
    return r.v;
}

__device__ __forceinline__ v8f wmma_bf16(v16bf a, v16bf b, v8f c) {
    return __builtin_amdgcn_wmma_f32_16x16x32_bf16(
        false, a, false, b, (short)0, c, false, false);
}

// CDNA5 async copy: global -> LDS, 8 bytes per lane, tracked by ASYNCcnt.
__device__ __forceinline__ void async_ld_lds_b64(void* lds, const void* g) {
    asm volatile("global_load_async_to_lds_b64 %0, %1, off"
                 :: "v"((uint32_t)(uintptr_t)lds),
                    "v"((uint64_t)(uintptr_t)g)
                 : "memory");
}
__device__ __forceinline__ void wait_async0() {
    asm volatile("s_wait_asynccnt 0x0" ::: "memory");
}

// ---------------------------------------------------------------------------
// fp32 -> bf16 conversion, 4 elements / thread
// ---------------------------------------------------------------------------
__global__ void mha_cvt_bf16(const float* __restrict__ src,
                             __bf16* __restrict__ dst, int n) {
    int i = (blockIdx.x * blockDim.x + threadIdx.x) * 4;
    if (i + 3 < n) {
        float4 v = *(const float4*)(src + i);
        dst[i + 0] = (__bf16)v.x;
        dst[i + 1] = (__bf16)v.y;
        dst[i + 2] = (__bf16)v.z;
        dst[i + 3] = (__bf16)v.w;
    } else {
        for (int k = i; k < n; ++k) dst[k] = (__bf16)src[k];
    }
}

// ---------------------------------------------------------------------------
// Shared GEMM main loop: 128x128 tile / WG (8 waves x 16-row strips),
// K-step 32, K-dim = E_DIM. Double-buffered LDS:
//   A tile: async global->LDS (row-major, stride 36 = 32+4 pad)
//   B tile: uint4 global loads issued before compute, transposed LDS
//           stores after compute (latency hidden behind WMMA).
// Fills acc[8] (8 column tiles x v8f) for this wave's 16-row strip.
// ---------------------------------------------------------------------------
#define LDSTR 36

__device__ __forceinline__ void gemm_main(
        const __bf16* __restrict__ X, const __bf16* __restrict__ W, int ldw,
        __bf16 (&A_lds)[2][128 * LDSTR], __bf16 (&B_lds)[2][128 * LDSTR],
        int m0, int col0, int tid, int wave, int g, int ln, v8f (&acc)[8]) {
#pragma unroll
    for (int t = 0; t < 8; ++t) acc[t] = (v8f){};
    uint4 breg[2];

    auto stage_A = [&](int buf, int k0) {
#pragma unroll
        for (int idx = tid; idx < 1024; idx += 256) {
            int r = idx >> 3, c = idx & 7;
            async_ld_lds_b64(&A_lds[buf][r * LDSTR + c * 4],
                             X + (size_t)(m0 + r) * E_DIM + k0 + c * 4);
        }
    };
    auto load_B = [&](int k0) {
#pragma unroll
        for (int i = 0; i < 2; ++i) {
            int idx = tid + i * 256;
            int k = idx >> 4, c = (idx & 15) * 8;
            breg[i] = *(const uint4*)(W + (size_t)(k0 + k) * ldw + col0 + c);
        }
    };
    auto store_B = [&](int buf) {
#pragma unroll
        for (int i = 0; i < 2; ++i) {
            int idx = tid + i * 256;
            int k = idx >> 4, c = (idx & 15) * 8;
            const __bf16* e = (const __bf16*)&breg[i];
#pragma unroll
            for (int j = 0; j < 8; ++j)
                B_lds[buf][(c + j) * LDSTR + k] = e[j];
        }
    };

    stage_A(0, 0); load_B(0); store_B(0);
    wait_async0(); __syncthreads();

    int cur = 0;
    for (int k0 = 0; k0 < E_DIM; k0 += 32) {
        const int nxt = cur ^ 1;
        const bool more = (k0 + 32) < E_DIM;
        if (more) { stage_A(nxt, k0 + 32); load_B(k0 + 32); }

        v16bf af = ld_frag32(
            (const uint32_t*)&A_lds[cur][(wave * 16 + ln) * LDSTR], g);
        v16bf bf[8];
#pragma unroll
        for (int t = 0; t < 8; ++t)
            bf[t] = ld_frag32(
                (const uint32_t*)&B_lds[cur][(t * 16 + ln) * LDSTR], g);
#pragma unroll
        for (int t = 0; t < 8; ++t) acc[t] = wmma_bf16(af, bf[t], acc[t]);

        if (more) store_B(nxt);
        wait_async0();
        __syncthreads();
        cur = nxt;
    }
}

// QKV GEMM: scatter into Q/K/V [b,h,n,d] decoding the (h,d,3) interleave.
__global__ void __launch_bounds__(256)
mha_qkv_gemm(const __bf16* __restrict__ X, const __bf16* __restrict__ W,
             const float* __restrict__ bias,
             __bf16* __restrict__ Qd, __bf16* __restrict__ Kd,
             __bf16* __restrict__ Vd) {
    __shared__ __bf16 A_lds[2][128 * LDSTR];
    __shared__ __bf16 B_lds[2][128 * LDSTR];
    const int tid  = threadIdx.x;
    const int lane = tid & 31, wave = tid >> 5;
    const int g = lane >> 4, ln = lane & 15;
    const int col0 = blockIdx.x * 128;
    const int m0   = blockIdx.y * 128;

    v8f acc[8];
    gemm_main(X, W, QKV_N, A_lds, B_lds, m0, col0, tid, wave, g, ln, acc);

#pragma unroll
    for (int t = 0; t < 8; ++t) {
        const int colg = col0 + t * 16 + ln;
        const int h = colg / (3 * D_HEAD);
        const int rem = colg % (3 * D_HEAD);
        const int d = rem / 3;
        const int which = rem % 3;
        const float bv = bias[colg];
#pragma unroll
        for (int j = 0; j < 8; ++j) {
            const int mg  = m0 + wave * 16 + j + 8 * g;
            const int b   = mg >> 11;
            const int row = mg & 2047;
            const size_t idx =
                (((size_t)(b * H_HEADS + h)) * N_SEQ + row) * D_HEAD + d;
            __bf16 ov = (__bf16)(acc[t][j] + bv);
            if (which == 0)      Qd[idx] = ov;
            else if (which == 1) Kd[idx] = ov;
            else                 Vd[idx] = ov;
        }
    }
}

// Proj GEMM: fp32 output + bias.
__global__ void __launch_bounds__(256)
mha_proj_gemm(const __bf16* __restrict__ X, const __bf16* __restrict__ W,
              const float* __restrict__ bias, float* __restrict__ out) {
    __shared__ __bf16 A_lds[2][128 * LDSTR];
    __shared__ __bf16 B_lds[2][128 * LDSTR];
    const int tid  = threadIdx.x;
    const int lane = tid & 31, wave = tid >> 5;
    const int g = lane >> 4, ln = lane & 15;
    const int col0 = blockIdx.x * 128;
    const int m0   = blockIdx.y * 128;

    v8f acc[8];
    gemm_main(X, W, E_DIM, A_lds, B_lds, m0, col0, tid, wave, g, ln, acc);

#pragma unroll
    for (int t = 0; t < 8; ++t) {
        const int colg = col0 + t * 16 + ln;
        const float bv = bias[colg];
#pragma unroll
        for (int j = 0; j < 8; ++j) {
            const int mg = m0 + wave * 16 + j + 8 * g;
            out[(size_t)mg * E_DIM + colg] = acc[t][j] + bv;
        }
    }
}

// ---------------------------------------------------------------------------
// Flash attention: grid (N/128, B*H). 8 waves x 16 q-rows, 64-key tiles,
// online softmax, scale 1/sqrt(E)=1/32. Double-buffered K/V LDS with async
// K staging and split-load V transpose.
// ---------------------------------------------------------------------------
#define KSTR 68

__global__ void __launch_bounds__(256)
mha_flash(const __bf16* __restrict__ Q, const __bf16* __restrict__ K,
          const __bf16* __restrict__ V, __bf16* __restrict__ O) {
    __shared__ __bf16 K_lds[2][64 * KSTR];       // row-major [key][d]
    __shared__ __bf16 V_lds[2][64 * KSTR];       // col-major [d][key]
    __shared__ __bf16 P_lds[8 * 16 * KSTR];      // per-wave [row][key]

    const int tid = threadIdx.x;
    const int lane = tid & 31, wave = tid >> 5;
    const int g = lane >> 4, ln = lane & 15;
    const int bh = blockIdx.y;
    const int q0 = blockIdx.x * 128;
    const size_t base = (size_t)bh * N_SEQ * D_HEAD;

    const int qrow = q0 + wave * 16 + ln;
    const uint32_t* qp = (const uint32_t*)(Q + base + (size_t)qrow * D_HEAD);
    v16bf qf0 = ld_frag32(qp, g);
    v16bf qf1 = ld_frag32(qp + 16, g);

    float m_run[8], l_run[8];
    v8f o[4];
#pragma unroll
    for (int j = 0; j < 8; ++j) { m_run[j] = -INFINITY; l_run[j] = 0.f; }
#pragma unroll
    for (int t = 0; t < 4; ++t) o[t] = (v8f){};

    const float SCALE = 0.03125f;                // 1/sqrt(1024)
    const float LOG2E = 1.44269504088896340736f;

    uint4 vreg[2];
    auto stage_K = [&](int buf, int j0) {
#pragma unroll
        for (int idx = tid; idx < 1024; idx += 256) {
            int r = idx >> 4, c = idx & 15;
            async_ld_lds_b64(&K_lds[buf][r * KSTR + c * 4],
                             K + base + (size_t)(j0 + r) * D_HEAD + c * 4);
        }
    };
    auto load_V = [&](int j0) {
#pragma unroll
        for (int i = 0; i < 2; ++i) {
            int idx = tid + i * 256;
            int key = idx >> 3, d0 = (idx & 7) * 8;
            vreg[i] = *(const uint4*)(V + base + (size_t)(j0 + key) * D_HEAD + d0);
        }
    };
    auto store_V = [&](int buf) {
#pragma unroll
        for (int i = 0; i < 2; ++i) {
            int idx = tid + i * 256;
            int key = idx >> 3, d0 = (idx & 7) * 8;
            const __bf16* e = (const __bf16*)&vreg[i];
#pragma unroll
            for (int j = 0; j < 8; ++j)
                V_lds[buf][(d0 + j) * KSTR + key] = e[j];
        }
    };

    stage_K(0, 0); load_V(0); store_V(0);
    wait_async0(); __syncthreads();

    int cur = 0;
    for (int j0 = 0; j0 < N_SEQ; j0 += 64) {
        const int nxt = cur ^ 1;
        const bool more = (j0 + 64) < N_SEQ;
        if (more) { stage_K(nxt, j0 + 64); load_V(j0 + 64); }

        // S = Q K^T : load all 8 B-frags, then 8 back-to-back wmma
        v16bf kf[8];
#pragma unroll
        for (int t = 0; t < 4; ++t) {
            const uint32_t* kp =
                (const uint32_t*)&K_lds[cur][(t * 16 + ln) * KSTR];
            kf[2 * t]     = ld_frag32(kp, g);
            kf[2 * t + 1] = ld_frag32(kp + 16, g);
        }
        v8f s[4];
#pragma unroll
        for (int t = 0; t < 4; ++t) {
            v8f z = (v8f){};
            z    = wmma_bf16(qf0, kf[2 * t], z);
            s[t] = wmma_bf16(qf1, kf[2 * t + 1], z);
        }

        // online softmax (rows j+8g; 16 lanes share a row)
#pragma unroll
        for (int j = 0; j < 8; ++j) {
            float mx = s[0][j];
#pragma unroll
            for (int t = 1; t < 4; ++t) mx = fmaxf(mx, s[t][j]);
            mx *= SCALE;
#pragma unroll
            for (int msk = 1; msk <= 8; msk <<= 1)
                mx = fmaxf(mx, __shfl_xor(mx, msk, 32));
            float mn = fmaxf(m_run[j], mx);
            float rs = 0.f;
#pragma unroll
            for (int t = 0; t < 4; ++t) {
                float p = exp2f((s[t][j] * SCALE - mn) * LOG2E);
                s[t][j] = p;
                rs += p;
            }
#pragma unroll
            for (int msk = 1; msk <= 8; msk <<= 1)
                rs += __shfl_xor(rs, msk, 32);
            float corr = exp2f((m_run[j] - mn) * LOG2E);
            l_run[j] = l_run[j] * corr + rs;
            m_run[j] = mn;
#pragma unroll
            for (int t = 0; t < 4; ++t) o[t][j] *= corr;
        }

        // C-layout -> A-layout via per-wave LDS strip
        __bf16* pb = &P_lds[wave * 16 * KSTR];
#pragma unroll
        for (int t = 0; t < 4; ++t)
#pragma unroll
            for (int j = 0; j < 8; ++j)
                pb[(j + 8 * g) * KSTR + t * 16 + ln] = (__bf16)s[t][j];
        asm volatile("s_wait_dscnt 0" ::: "memory");

        const uint32_t* pp =
            (const uint32_t*)&P_lds[(wave * 16 + ln) * KSTR];
        v16bf pf0 = ld_frag32(pp, g);
        v16bf pf1 = ld_frag32(pp + 16, g);

        // O += P V : load all V-frags then 8 wmma
        v16bf vf[8];
#pragma unroll
        for (int t = 0; t < 4; ++t) {
            const uint32_t* vp =
                (const uint32_t*)&V_lds[cur][(t * 16 + ln) * KSTR];
            vf[2 * t]     = ld_frag32(vp, g);
            vf[2 * t + 1] = ld_frag32(vp + 16, g);
        }
#pragma unroll
        for (int t = 0; t < 4; ++t) {
            o[t] = wmma_bf16(pf0, vf[2 * t], o[t]);
            o[t] = wmma_bf16(pf1, vf[2 * t + 1], o[t]);
        }

        if (more) store_V(nxt);
        wait_async0();
        __syncthreads();
        cur = nxt;
    }

    // normalize, store to [b, n, h*64+d] bf16 (feeds proj GEMM)
    const int b = bh >> 4, h = bh & 15;
#pragma unroll
    for (int j = 0; j < 8; ++j) {
        const float inv = 1.0f / l_run[j];
        const int row = q0 + wave * 16 + j + 8 * g;
        const size_t rbase = ((size_t)(b * N_SEQ + row)) * E_DIM + h * D_HEAD;
#pragma unroll
        for (int t = 0; t < 4; ++t)
            O[rbase + t * 16 + ln] = (__bf16)(o[t][j] * inv);
    }
}

// ---------------------------------------------------------------------------
extern "C" void kernel_launch(void* const* d_in, const int* in_sizes, int n_in,
                              void* d_out, int out_size, void* d_ws, size_t ws_size,
                              hipStream_t stream) {
    const float* x     = (const float*)d_in[0];
    const float* Wqkv  = (const float*)d_in[1];
    const float* bqkv  = (const float*)d_in[2];
    const float* Wproj = (const float*)d_in[3];
    const float* bproj = (const float*)d_in[4];
    float* out = (float*)d_out;

    char* ws = (char*)d_ws;
    size_t off = 0;
    __bf16* xb   = (__bf16*)(ws + off); off += (size_t)M_ROWS * E_DIM * 2;
    __bf16* wqb  = (__bf16*)(ws + off); off += (size_t)E_DIM * QKV_N * 2;
    __bf16* wpb  = (__bf16*)(ws + off); off += (size_t)E_DIM * E_DIM * 2;
    __bf16* Qb   = (__bf16*)(ws + off); off += (size_t)M_ROWS * E_DIM * 2;
    __bf16* Kb   = (__bf16*)(ws + off); off += (size_t)M_ROWS * E_DIM * 2;
    __bf16* Vb   = (__bf16*)(ws + off); off += (size_t)M_ROWS * E_DIM * 2;
    __bf16* attb = (__bf16*)(ws + off); off += (size_t)M_ROWS * E_DIM * 2;
    (void)ws_size; (void)in_sizes; (void)n_in; (void)out_size;

    const int nx = M_ROWS * E_DIM;
    const int nw = E_DIM * QKV_N;
    const int np = E_DIM * E_DIM;
    mha_cvt_bf16<<<(nx / 4 + 255) / 256, 256, 0, stream>>>(x, xb, nx);
    mha_cvt_bf16<<<(nw / 4 + 255) / 256, 256, 0, stream>>>(Wqkv, wqb, nw);
    mha_cvt_bf16<<<(np / 4 + 255) / 256, 256, 0, stream>>>(Wproj, wpb, np);

    mha_qkv_gemm<<<dim3(QKV_N / 128, M_ROWS / 128), 256, 0, stream>>>(
        xb, wqb, bqkv, Qb, Kb, Vb);

    mha_flash<<<dim3(N_SEQ / 128, B_SZ * H_HEADS), 256, 0, stream>>>(
        Qb, Kb, Vb, attb);

    mha_proj_gemm<<<dim3(E_DIM / 128, M_ROWS / 128), 256, 0, stream>>>(
        attb, wpb, bproj, out);
}
